// DN_21758304321876
// MI455X (gfx1250) — matrix operand order repaired and checked by reading, not compiled, not run on hardware
//
#include <hip/hip_runtime.h>

// ---------------------------------------------------------------------------
// DN forward (test mode) for MI455X / gfx1250, wave32 + WMMA bf16.
//   out[b,z] = y2z_w[z, argmax_y( gate(y) * <xhat_b, w_xy_y / |w_xy_y|> )] / |y2z_w_z|
// ---------------------------------------------------------------------------

typedef __attribute__((ext_vector_type(16))) __bf16        v16bf;
typedef __attribute__((ext_vector_type(8)))  float         v8f;
typedef __attribute__((ext_vector_type(4)))  unsigned int  v4u;
typedef __attribute__((ext_vector_type(4)))  float         v4f;

union FragBF { v16bf bf; v4u u[2]; };

#define XDIM   4096
#define BDIM   512
#define YDIM   32768
#define ZDIM   1000
#define BN     64         // y rows per block in the fused GEMM
#define BPITCH 36         // LDS B-tile row pitch in dwords (32 + 4 pad -> no bank conflicts)
#define NCHUNK (XDIM / 64)
#define EPSN   1e-12f

// round-to-nearest-even fp32 -> bf16, packed pair into one dword
static __device__ __forceinline__ unsigned int pack_bf16(float a, float b) {
  unsigned int ua = __float_as_uint(a);
  unsigned int ub = __float_as_uint(b);
  ua += 0x7FFFu + ((ua >> 16) & 1u);
  ub += 0x7FFFu + ((ub >> 16) & 1u);
  return (ua >> 16) | (ub & 0xFFFF0000u);
}

// monotone map fp32 -> u32 (order preserving for atomicMax)
static __device__ __forceinline__ unsigned int fkey(float f) {
  unsigned int u = __float_as_uint(f);
  return (u & 0x80000000u) ? ~u : (u | 0x80000000u);
}

// swap bits 2<->3 of a dword-index within a 16-dword (32 bf16) chunk:
// this is the WMMA 16-bit A-operand k-swizzle (elements: swap bits 3<->4).
static __device__ __forceinline__ int swz(int d) {
  return (d & ~12) | ((d & 4) << 1) | ((d & 8) >> 1);
}

// ---------------------------------------------------------------------------
// K1: xf = l2norm(x rows), stored as bf16 pre-swizzled for WMMA A fragments.
// One block per batch row.
// ---------------------------------------------------------------------------
__global__ __launch_bounds__(256) void normx_kernel(const float* __restrict__ x,
                                                    unsigned int* __restrict__ xf) {
  __shared__ float red[256];
  const int b = blockIdx.x, t = threadIdx.x;
  const v4f* row = (const v4f*)(x + (size_t)b * XDIM);
  v4f v0 = row[t * 4 + 0], v1 = row[t * 4 + 1], v2 = row[t * 4 + 2], v3 = row[t * 4 + 3];
  float s = 0.f;
#pragma unroll
  for (int j = 0; j < 4; ++j) { s += v0[j]*v0[j] + v1[j]*v1[j] + v2[j]*v2[j] + v3[j]*v3[j]; }
  red[t] = s;
  __syncthreads();
  for (int off = 128; off > 0; off >>= 1) {
    if (t < off) red[t] += red[t + off];
    __syncthreads();
  }
  const float inv = 1.0f / fmaxf(sqrtf(red[0]), EPSN);
  float e[16];
#pragma unroll
  for (int j = 0; j < 4; ++j) { e[j] = v0[j]*inv; e[4+j] = v1[j]*inv; e[8+j] = v2[j]*inv; e[12+j] = v3[j]*inv; }
  unsigned int d[8];
#pragma unroll
  for (int j = 0; j < 8; ++j) d[j] = pack_bf16(e[2*j], e[2*j+1]);
  // thread owns row-dwords [8t, 8t+8); 4-aligned groups stay contiguous under swz
  const int base  = b * (XDIM / 2);
  const int D0    = 8 * t;
  const int chunk = D0 & ~15;
  const int p0    = chunk + swz(D0 & 15);
  const int p1    = chunk + swz((D0 & 15) + 4);
  *(v4u*)(xf + base + p0) = (v4u){d[0], d[1], d[2], d[3]};
  *(v4u*)(xf + base + p1) = (v4u){d[4], d[5], d[6], d[7]};
}

// ---------------------------------------------------------------------------
// K2: inv_nz[z] = 1 / max(|y2z_w[z,:]|, eps).  One block per z row (streams 128MB).
// ---------------------------------------------------------------------------
__global__ __launch_bounds__(256) void y2znorm_kernel(const float* __restrict__ wz,
                                                      float* __restrict__ inv_nz) {
  __shared__ float red[256];
  const int z = blockIdx.x, t = threadIdx.x;
  const v4f* row = (const v4f*)(wz + (size_t)z * YDIM);
  float s = 0.f;
  for (int i = t; i < YDIM / 4; i += 256) {
    v4f f = __builtin_nontemporal_load(row + i);
    s += f[0]*f[0] + f[1]*f[1] + f[2]*f[2] + f[3]*f[3];
  }
  red[t] = s;
  __syncthreads();
  for (int off = 128; off > 0; off >>= 1) {
    if (t < off) red[t] += red[t + off];
    __syncthreads();
  }
  if (t == 0) inv_nz[z] = 1.0f / fmaxf(sqrtf(red[0]), EPSN);
}

// ---------------------------------------------------------------------------
// K3: fused GEMM (bf16 WMMA) + w-row norms + age gate + argmax-over-Y.
// Block: 64 y-rows x all 512 b-rows x K=4096.  256 threads = 8 waves.
// Wave layout 8x1 (M x N): wave owns 4 m-tiles (64 b rows) x all 4 n-tiles
//   -> 16 C tiles (128 accum VGPRs), 4 A frags + 4 B frags live per k-step.
// Software pipeline: double-buffered LDS B-tile, one barrier per K64 chunk;
// next chunk's HBM loads are issued before the 32 WMMAs of the current chunk.
// ---------------------------------------------------------------------------
__global__ __launch_bounds__(256) void gemm_argmax_kernel(
    const float* __restrict__ w,            // [Y, X] row-major fp32
    const unsigned int* __restrict__ xf,    // [B, X/2] swizzled bf16 pairs
    const int* __restrict__ age,            // [Y]
    unsigned long long* __restrict__ winner /* [B] packed (key<<32 | ~y) */) {
  __shared__ unsigned int        btile[2 * BN * BPITCH];  // double-buffered K64 chunk
  __shared__ float               ssq[256];
  __shared__ float               scale_s[BN];
  __shared__ unsigned long long  best[BDIM];

  const int t    = threadIdx.x;
  const int yb   = blockIdx.x * BN;
  best[t]       = 0ull;
  best[t + 256] = 0ull;

  const int wave = t >> 5, lane = t & 31;
  const int lrow = lane & 15, lg = lane >> 4;

  const v8f vzero = {0.f, 0.f, 0.f, 0.f, 0.f, 0.f, 0.f, 0.f};
  v8f acc[4][4];
#pragma unroll
  for (int mt = 0; mt < 4; ++mt)
#pragma unroll
    for (int nt = 0; nt < 4; ++nt) acc[mt][nt] = vzero;

  // cooperative B-slice staging: thread -> (row t/4, 16 floats at (t%4)*16)
  const int     brow = t >> 2;
  const int     bk16 = (t & 3) * 16;
  const float*  wrow = w + (size_t)(yb + brow) * XDIM + bk16;
  unsigned int* bst  = &btile[brow * BPITCH + (bk16 >> 1)];
  float         mysq = 0.f;

  // A fragment base (dwords): row = 64*wave + 16*mt + lrow, group offset 8*lg
  const unsigned int* abase =
      xf + (size_t)(64 * wave + lrow) * (XDIM / 2) + 8 * lg;

  // pipeline prologue: chunk 0 in flight
  v4f f0 = __builtin_nontemporal_load((const v4f*)(wrow));
  v4f f1 = __builtin_nontemporal_load((const v4f*)(wrow + 4));
  v4f f2 = __builtin_nontemporal_load((const v4f*)(wrow + 8));
  v4f f3 = __builtin_nontemporal_load((const v4f*)(wrow + 12));

#pragma unroll 1
  for (int c2 = 0; c2 < NCHUNK; ++c2) {
    // ---- commit current chunk to LDS buf[c2&1], fused sum-of-squares ----
    unsigned int* bp = bst + (c2 & 1) * (BN * BPITCH);
    *(v4u*)(bp)     = (v4u){pack_bf16(f0[0], f0[1]), pack_bf16(f0[2], f0[3]),
                            pack_bf16(f1[0], f1[1]), pack_bf16(f1[2], f1[3])};
    *(v4u*)(bp + 4) = (v4u){pack_bf16(f2[0], f2[1]), pack_bf16(f2[2], f2[3]),
                            pack_bf16(f3[0], f3[1]), pack_bf16(f3[2], f3[3])};
    mysq += f0[0]*f0[0] + f0[1]*f0[1] + f0[2]*f0[2] + f0[3]*f0[3]
          + f1[0]*f1[0] + f1[1]*f1[1] + f1[2]*f1[2] + f1[3]*f1[3]
          + f2[0]*f2[0] + f2[1]*f2[1] + f2[2]*f2[2] + f2[3]*f2[3]
          + f3[0]*f3[0] + f3[1]*f3[1] + f3[2]*f3[2] + f3[3]*f3[3];
    __syncthreads();   // waits dscnt first: buf[c2&1] globally visible

    // ---- issue next chunk's HBM loads NOW (latency hidden by the WMMAs) ----
    const int cn = (c2 + 1 < NCHUNK) ? (c2 + 1) : c2;   // clamp: harmless re-read
    f0 = __builtin_nontemporal_load((const v4f*)(wrow + cn * 64));
    f1 = __builtin_nontemporal_load((const v4f*)(wrow + cn * 64 + 4));
    f2 = __builtin_nontemporal_load((const v4f*)(wrow + cn * 64 + 8));
    f3 = __builtin_nontemporal_load((const v4f*)(wrow + cn * 64 + 12));
    __builtin_prefetch(wrow + (c2 + 2) * 64, 0, 1);     // speculative, 2 chunks ahead

    // ---- 2 k-steps x 16 WMMA from buf[c2&1] ----
    const unsigned int* bbuf = &btile[(c2 & 1) * (BN * BPITCH)];
#pragma unroll
    for (int s = 0; s < 2; ++s) {
      const int c = 2 * c2 + s;            // 32-wide k-step index
      FragBF a[4], bfr[4];
#pragma unroll
      for (int mt = 0; mt < 4; ++mt) {
        const unsigned int* ap = abase + (size_t)(16 * mt) * (XDIM / 2) + c * 16;
        a[mt].u[0] = *(const v4u*)(ap);
        a[mt].u[1] = *(const v4u*)(ap + 4);
      }
#pragma unroll
      for (int nt = 0; nt < 4; ++nt) {
        const unsigned int* bpp = bbuf + (16 * nt + lrow) * BPITCH + s * 16 + 4 * lg;
        bfr[nt].u[0] = *(const v4u*)(bpp);
        bfr[nt].u[1] = *(const v4u*)(bpp + 8);
      }
#pragma unroll
      for (int mt = 0; mt < 4; ++mt)
#pragma unroll
        for (int nt = 0; nt < 4; ++nt)
          acc[mt][nt] = __builtin_amdgcn_wmma_f32_16x16x32_bf16(
              false, a[mt].bf, false, bfr[nt].bf, (short)0, acc[mt][nt], false, false);
    }
    // no trailing barrier: next iteration stores to the other buffer; the
    // barrier at the top (with its dscnt wait) protects buffer reuse.
  }

  // ---- w row norms + age gate ----
  ssq[t] = mysq;
  __syncthreads();
  if (t < BN) {
    const float s   = ssq[4*t] + ssq[4*t+1] + ssq[4*t+2] + ssq[4*t+3];
    const float inv = 1.0f / fmaxf(sqrtf(s), EPSN);
    scale_s[t]      = (age[yb + t] >= 1) ? inv : 0.0f;
  }
  __syncthreads();

  // ---- epilogue: per-b argmax; combine n-tiles in registers, then LDS, then global
  float        sc[4];
  unsigned int yp[4];
#pragma unroll
  for (int nt = 0; nt < 4; ++nt) {
    const int ylocal = 16 * nt + lrow;
    sc[nt] = scale_s[ylocal];
    yp[nt] = 0xFFFFFFFFu - (unsigned int)(yb + ylocal);
  }
#pragma unroll
  for (int mt = 0; mt < 4; ++mt) {
#pragma unroll
    for (int r = 0; r < 8; ++r) {
      unsigned long long k = 0ull;
#pragma unroll
      for (int nt = 0; nt < 4; ++nt) {
        const unsigned long long key =
            ((unsigned long long)fkey(acc[mt][nt][r] * sc[nt]) << 32) | yp[nt];
        k = (key > k) ? key : k;
      }
      const int m = 64 * wave + 16 * mt + r + 8 * lg;   // C layout: M = r + 8*(lane/16)
      atomicMax(&best[m], k);
    }
  }
  __syncthreads();
  atomicMax(&winner[t],       best[t]);
  atomicMax(&winner[t + 256], best[t + 256]);
}

// ---------------------------------------------------------------------------
// K4: out[b,z] = y2z_w[z, winner_b] * inv_nz[z]
// ---------------------------------------------------------------------------
__global__ __launch_bounds__(256) void gather_kernel(
    const float* __restrict__ wz, const unsigned long long* __restrict__ winner,
    const float* __restrict__ inv_nz, float* __restrict__ out) {
  const int i = blockIdx.x * 256 + threadIdx.x;
  if (i >= BDIM * ZDIM) return;
  const int b = i / ZDIM;
  const int z = i - b * ZDIM;
  const unsigned int y = 0xFFFFFFFFu - (unsigned int)(winner[b] & 0xFFFFFFFFull);
  out[i] = wz[(size_t)z * YDIM + y] * inv_nz[z];
}

// ---------------------------------------------------------------------------
extern "C" void kernel_launch(void* const* d_in, const int* in_sizes, int n_in,
                              void* d_out, int out_size, void* d_ws, size_t ws_size,
                              hipStream_t stream) {
  (void)in_sizes; (void)n_in; (void)out_size; (void)ws_size;
  const float* x    = (const float*)d_in[0];   // [512,64,64]
  const float* w_xy = (const float*)d_in[2];   // [32768,4096]
  const float* w_yz = (const float*)d_in[3];   // [1000,32768]
  const int*   age  = (const int*)d_in[4];     // [1,32768]

  char* ws = (char*)d_ws;
  unsigned int*       xf     = (unsigned int*)ws;                          // 4 MB bf16 xhat (swizzled)
  unsigned long long* winner = (unsigned long long*)(ws + (size_t)BDIM * (XDIM / 2) * 4);
  float*              inv_nz = (float*)((char*)winner + BDIM * 8);

  hipMemsetAsync(winner, 0, BDIM * 8, stream);
  normx_kernel<<<BDIM, 256, 0, stream>>>(x, xf);
  y2znorm_kernel<<<ZDIM, 256, 0, stream>>>(w_yz, inv_nz);
  gemm_argmax_kernel<<<YDIM / BN, 256, 0, stream>>>(w_xy, xf, age, winner);
  gather_kernel<<<(BDIM * ZDIM + 255) / 256, 256, 0, stream>>>(w_yz, winner, inv_nz, (float*)d_out);
}
